// RWKVModel_69861938036989
// MI455X (gfx1250) — matrix operand order repaired and checked by reading, not compiled, not run on hardware
//
#include <hip/hip_runtime.h>
#include <hip/hip_bf16.h>

// ---------------- types / helpers ----------------
typedef __attribute__((ext_vector_type(16))) __bf16       v16bf;
typedef __attribute__((ext_vector_type(8)))  float        v8f;
typedef __attribute__((ext_vector_type(4)))  unsigned int u32x4;

union FragBF { v16bf v; u32x4 q[2]; };

__device__ __forceinline__ unsigned short f32_to_bf16(float f) {
  unsigned int u = __float_as_uint(f);
  unsigned int r = 0x7FFFu + ((u >> 16) & 1u);   // round-to-nearest-even
  return (unsigned short)((u + r) >> 16);
}

static constexpr int Bc = 2, Sc = 512, Mc = 1024, Dc = 768, Hc = 3072, Vc = 100300, Vpc = 100352;

// epilogue modes
static constexpr int EPI_F32 = 0, EPI_SIGMOID = 1, EPI_RELUSQ = 2, EPI_ADD = 3, EPI_MULADD = 4,
                     EPI_F32_NT = 5;

// ---------------- weight fp32 -> bf16 pre-swizzle into WMMA B-fragment layout ----------------
// W is [K,N] row-major fp32. dst fragments: [nt][kt][lane][16 bf16], where for
// 32x16 bf16 B tiles: lane = n_in | ((k_in>>4)<<4), element e = k_in & 15.
__global__ void k_swizzle(const float* __restrict__ W, unsigned short* __restrict__ dst,
                          int K, int N, int Npad) {
  size_t i = (size_t)blockIdx.x * 256 + threadIdx.x;
  size_t total = (size_t)K * (size_t)Npad;
  if (i >= total) return;
  int n = (int)(i % Npad);
  int k = (int)(i / Npad);
  int kt = k >> 5, k_in = k & 31;
  int nt = n >> 4, n_in = n & 15;
  int lane = n_in | ((k_in >> 4) << 4);
  int e = k_in & 15;
  int KT = K >> 5;
  size_t didx = (((size_t)nt * KT + kt) * 32 + lane) * 16 + e;
  float v = (n < N) ? W[(size_t)k * N + n] : 0.0f;
  dst[didx] = f32_to_bf16(v);
}

// ---------------- layernorm (optionally gathering embedding rows) ----------------
// one block per row (256 threads, 3 elements each, D=768)
template <bool BF16OUT>
__global__ void k_ln(const int* __restrict__ tokens, const float* __restrict__ xbase,
                     const float* __restrict__ g, const float* __restrict__ b,
                     float* __restrict__ of, unsigned short* __restrict__ oh) {
  __shared__ float red[256];
  const int m = blockIdx.x;
  const int t = threadIdx.x;
  size_t srcRow = tokens ? (size_t)tokens[m] : (size_t)m;
  const float* row = xbase + srcRow * Dc;
  float v[3];
  float s = 0.f;
  #pragma unroll
  for (int i = 0; i < 3; ++i) { v[i] = row[t + i * 256]; s += v[i]; }
  red[t] = s; __syncthreads();
  for (int off = 128; off > 0; off >>= 1) { if (t < off) red[t] += red[t + off]; __syncthreads(); }
  const float mean = red[0] * (1.0f / Dc);
  __syncthreads();
  float s2 = 0.f;
  #pragma unroll
  for (int i = 0; i < 3; ++i) { float d = v[i] - mean; s2 += d * d; }
  red[t] = s2; __syncthreads();
  for (int off = 128; off > 0; off >>= 1) { if (t < off) red[t] += red[t + off]; __syncthreads(); }
  const float rstd = rsqrtf(red[0] * (1.0f / Dc) + 1e-5f);
  #pragma unroll
  for (int i = 0; i < 3; ++i) {
    int d = t + i * 256;
    float o = (v[i] - mean) * rstd * g[d] + b[d];
    if (BF16OUT) oh[(size_t)m * Dc + d] = f32_to_bf16(o);
    else         of[(size_t)m * Dc + d] = o;
  }
}

// ---------------- token-shift mixes (elementwise), bf16 outputs ----------------
__global__ void k_tm_mix(const float* __restrict__ xx, const float* __restrict__ muk,
                         const float* __restrict__ muv, const float* __restrict__ mur,
                         unsigned short* __restrict__ ak, unsigned short* __restrict__ av,
                         unsigned short* __restrict__ ar) {
  const int d = blockIdx.x * 256 + threadIdx.x;   // gridDim.x == 3
  const int m = blockIdx.y;
  const size_t i = (size_t)m * Dc + d;
  const float cur = xx[i];
  const float prev = (m & (Sc - 1)) ? xx[i - Dc] : 0.0f;
  const float mk = muk[d], mv = muv[d], mr = mur[d];
  ak[i] = f32_to_bf16(cur * mk + prev * (1.0f - mk));
  av[i] = f32_to_bf16(cur * mv + prev * (1.0f - mv));
  ar[i] = f32_to_bf16(cur * mr + prev * (1.0f - mr));
}

__global__ void k_cm_mix(const float* __restrict__ xx, const float* __restrict__ muk,
                         const float* __restrict__ mur,
                         unsigned short* __restrict__ ak, unsigned short* __restrict__ ar) {
  const int d = blockIdx.x * 256 + threadIdx.x;
  const int m = blockIdx.y;
  const size_t i = (size_t)m * Dc + d;
  const float cur = xx[i];
  const float prev = (m & (Sc - 1)) ? xx[i - Dc] : 0.0f;
  const float mk = muk[d], mr = mur[d];
  ak[i] = f32_to_bf16(cur * mk + prev * (1.0f - mk));
  ar[i] = f32_to_bf16(cur * mr + prev * (1.0f - mr));
}

// ---------------- WKV max-stabilized scan; writes bf16(r * wkv) for the Wo GEMM ----------------
__global__ void k_wkv(const float* __restrict__ kb, const float* __restrict__ vb,
                      const float* __restrict__ w, const float* __restrict__ u,
                      const float* __restrict__ r, unsigned short* __restrict__ aatt) {
  const int i = blockIdx.x * 256 + threadIdx.x;   // over B*D = 1536 channels
  if (i >= Bc * Dc) return;
  const int bidx = i / Dc, d = i - bidx * Dc;
  const float decay = -__expf(w[d]);
  const float uu = u[d];
  float p = -1e38f, a = 0.0f, bbv = 0.0f;
  const size_t base = (size_t)bidx * Sc * Dc + d;
  for (int t = 0; t < Sc; ++t) {
    const size_t idx = base + (size_t)t * Dc;
    const float kt = kb[idx], vt = vb[idx];
    float ww = uu + kt;
    float q = fmaxf(p, ww);
    float e1 = __expf(p - q), e2 = __expf(ww - q);
    float out = (e1 * a + e2 * vt) / (e1 * bbv + e2);
    float ww2 = p + decay;
    float q2 = fmaxf(ww2, kt);
    e1 = __expf(ww2 - q2); e2 = __expf(kt - q2);
    p = q2; a = e1 * a + e2 * vt; bbv = e1 * bbv + e2;
    aatt[idx] = f32_to_bf16(r[idx] * out);
  }
}

// ---------------- WMMA bf16 GEMM (double-buffered, branch-free inner loop) ----------------
// C[M,N] = A[M,K] (bf16 row-major) x Wswz (bf16 pre-swizzled [nt][kt][lane][16])
// block = 256 threads = 8 waves; wave = one 16-row m-tile x four 16-col n-tiles.
// KT = K/32 is even for every GEMM here (24 or 96), so a 2-deep pipeline maps exactly.
template <int EPI>
__global__ void __launch_bounds__(256)
k_gemm(const unsigned short* __restrict__ A, const unsigned short* __restrict__ W,
       float* __restrict__ outF, unsigned short* __restrict__ outH,
       const float* __restrict__ aux, int N, int Npad, int K) {
  const int lane = threadIdx.x & 31;
  const int wave = threadIdx.x >> 5;
  const int mt = blockIdx.y;
  const int nt0 = blockIdx.x * 32 + wave * 4;        // first 16-col tile of this wave
  if (nt0 * 16 >= Npad) return;                      // wave-uniform early out (EXEC stays full)
  const int KT = K >> 5;
  const int m = mt * 16 + (lane & 15);
  const int kb = (lane >> 4) << 4;                   // 0 or 16: A-layout K base per half-wave
  const unsigned short* Arow = A + (size_t)m * K + kb;
  const unsigned short* Wl = W + (size_t)nt0 * KT * 512 + (size_t)lane * 16;

  v8f acc[4] = {};
  FragBF aF[2];
  FragBF bF[2][4];

  auto loadStage = [&](int kt, int buf) {
    const unsigned short* ap = Arow + kt * 32;
    aF[buf].q[0] = *(const u32x4*)(ap);               // K = kb..kb+7
    aF[buf].q[1] = *(const u32x4*)(ap + 8);           // K = kb+8..kb+15
    #pragma unroll
    for (int t = 0; t < 4; ++t) {
      const unsigned short* pw = Wl + ((size_t)t * KT + kt) * 512;
      bF[buf][t].q[0] = *(const u32x4*)(pw);
      bF[buf][t].q[1] = *(const u32x4*)(pw + 8);
      __builtin_prefetch(pw + 1024, 0, 1);            // distance-2 fragment prefetch (speculative)
    }
  };
  auto mmaStage = [&](int buf) {
    #pragma unroll
    for (int t = 0; t < 4; ++t)
      acc[t] = __builtin_amdgcn_wmma_f32_16x16x32_bf16(
          false, aF[buf].v, false, bF[buf][t].v, (short)0, acc[t], false, false);
  };

  loadStage(0, 0);
  for (int kt = 0; kt < KT; kt += 2) {
    loadStage(kt + 1, 1);                             // overlap with mmaStage(0)
    mmaStage(0);
    const int kn = (kt + 2 < KT) ? (kt + 2) : 0;      // branch-free clamp; last reload is dead
    loadStage(kn, 0);                                 // overlap with mmaStage(1)
    mmaStage(1);
  }

  // epilogue: lane holds col = nt*16 + (lane&15); acc[j] -> row = mt*16 + j + ((lane>>4)<<3)
  const int row0 = mt * 16 + ((lane >> 4) << 3);
  const int col = lane & 15;
  #pragma unroll
  for (int t = 0; t < 4; ++t) {
    const int n = (nt0 + t) * 16 + col;
    if (n >= N) continue;
    #pragma unroll
    for (int j = 0; j < 8; ++j) {
      const size_t idx = (size_t)(row0 + j) * N + n;
      const float v = acc[t][j];
      if constexpr (EPI == EPI_F32)         outF[idx] = v;
      else if constexpr (EPI == EPI_F32_NT) __builtin_nontemporal_store(v, &outF[idx]);
      else if constexpr (EPI == EPI_SIGMOID) outF[idx] = 1.0f / (1.0f + __expf(-v));
      else if constexpr (EPI == EPI_RELUSQ) { float rr = v > 0.0f ? v : 0.0f; outH[idx] = f32_to_bf16(rr * rr); }
      else if constexpr (EPI == EPI_ADD)    outF[idx] += v;
      else if constexpr (EPI == EPI_MULADD) outF[idx] += aux[idx] * v;
    }
  }
}

// ---------------- host launcher ----------------
extern "C" void kernel_launch(void* const* d_in, const int* in_sizes, int n_in,
                              void* d_out, int out_size, void* d_ws, size_t ws_size,
                              hipStream_t stream) {
  (void)in_sizes; (void)n_in; (void)out_size; (void)ws_size;
  const int*   tokens  = (const int*)  d_in[0];
  const float* emb     = (const float*)d_in[1];
  const float* ln0_g   = (const float*)d_in[2];
  const float* ln0_b   = (const float*)d_in[3];
  const float* ln1_g   = (const float*)d_in[4];
  const float* ln1_b   = (const float*)d_in[5];
  const float* ln2_g   = (const float*)d_in[6];
  const float* ln2_b   = (const float*)d_in[7];
  const float* tdec    = (const float*)d_in[8];
  const float* tfirst  = (const float*)d_in[9];
  const float* tm_mu_k = (const float*)d_in[10];
  const float* tm_mu_v = (const float*)d_in[11];
  const float* tm_mu_r = (const float*)d_in[12];
  const float* tm_Wk   = (const float*)d_in[13];
  const float* tm_Wv   = (const float*)d_in[14];
  const float* tm_Wr   = (const float*)d_in[15];
  const float* tm_Wo   = (const float*)d_in[16];
  const float* cm_mu_k = (const float*)d_in[17];
  const float* cm_mu_r = (const float*)d_in[18];
  const float* cm_Wk   = (const float*)d_in[19];
  const float* cm_Wv   = (const float*)d_in[20];
  const float* cm_Wr   = (const float*)d_in[21];
  const float* lnf_g   = (const float*)d_in[22];
  const float* lnf_b   = (const float*)d_in[23];
  const float* headW   = (const float*)d_in[24];

  char* p = (char*)d_ws;
  auto carve = [&](size_t bytes) -> char* {
    char* r = p; p += (bytes + 255) & ~(size_t)255; return r;
  };

  const size_t DD = (size_t)Dc * Dc, DH = (size_t)Dc * Hc;
  unsigned short *sWr[12], *sWk[12], *sWv[12], *sWo[12], *sCk[12], *sCv[12], *sCr[12];
  for (int l = 0; l < 12; ++l) {
    sWr[l] = (unsigned short*)carve(DD * 2);
    sWk[l] = (unsigned short*)carve(DD * 2);
    sWv[l] = (unsigned short*)carve(DD * 2);
    sWo[l] = (unsigned short*)carve(DD * 2);
    sCk[l] = (unsigned short*)carve(DH * 2);
    sCv[l] = (unsigned short*)carve(DH * 2);
    sCr[l] = (unsigned short*)carve(DD * 2);
  }
  unsigned short* sHead = (unsigned short*)carve((size_t)Dc * Vpc * 2);

  float* x     = (float*)carve((size_t)Mc * Dc * 4);
  float* xx    = (float*)carve((size_t)Mc * Dc * 4);
  unsigned short* axk  = (unsigned short*)carve((size_t)Mc * Dc * 2);
  unsigned short* axv  = (unsigned short*)carve((size_t)Mc * Dc * 2);
  unsigned short* axr  = (unsigned short*)carve((size_t)Mc * Dc * 2);
  float* rbuf  = (float*)carve((size_t)Mc * Dc * 4);
  float* kbuf  = (float*)carve((size_t)Mc * Dc * 4);
  float* vbuf  = (float*)carve((size_t)Mc * Dc * 4);
  unsigned short* aatt = (unsigned short*)carve((size_t)Mc * Dc * 2);
  unsigned short* kkh  = (unsigned short*)carve((size_t)Mc * Hc * 2);
  float* r2    = (float*)carve((size_t)Mc * Dc * 4);
  unsigned short* ahead= (unsigned short*)carve((size_t)Mc * Dc * 2);

  auto swz = [&](const float* W, unsigned short* dst, int K, int N, int Npad) {
    size_t total = (size_t)K * Npad;
    k_swizzle<<<(unsigned)((total + 255) / 256), 256, 0, stream>>>(W, dst, K, N, Npad);
  };
  for (int l = 0; l < 12; ++l) {
    swz(tm_Wr + (size_t)l * DD, sWr[l], Dc, Dc, Dc);
    swz(tm_Wk + (size_t)l * DD, sWk[l], Dc, Dc, Dc);
    swz(tm_Wv + (size_t)l * DD, sWv[l], Dc, Dc, Dc);
    swz(tm_Wo + (size_t)l * DD, sWo[l], Dc, Dc, Dc);
    swz(cm_Wk + (size_t)l * DH, sCk[l], Dc, Hc, Hc);
    swz(cm_Wv + (size_t)l * DH, sCv[l], Hc, Dc, Dc);
    swz(cm_Wr + (size_t)l * DD, sCr[l], Dc, Dc, Dc);
  }
  swz(headW, sHead, Dc, Vc, Vpc);

  auto gGrid = [&](int Npad) { return dim3((unsigned)((Npad + 511) / 512), Mc / 16); };

  // embedding gather + ln0 -> residual stream x
  k_ln<false><<<Mc, 256, 0, stream>>>(tokens, emb, ln0_g, ln0_b, x, nullptr);

  for (int l = 0; l < 12; ++l) {
    // ---- time mixing ----
    k_ln<false><<<Mc, 256, 0, stream>>>(nullptr, x, ln1_g + l * Dc, ln1_b + l * Dc, xx, nullptr);
    k_tm_mix<<<dim3(3, Mc), 256, 0, stream>>>(xx, tm_mu_k + l * Dc, tm_mu_v + l * Dc,
                                              tm_mu_r + l * Dc, axk, axv, axr);
    k_gemm<EPI_SIGMOID><<<gGrid(Dc), 256, 0, stream>>>(axr, sWr[l], rbuf, nullptr, nullptr, Dc, Dc, Dc);
    k_gemm<EPI_F32    ><<<gGrid(Dc), 256, 0, stream>>>(axk, sWk[l], kbuf, nullptr, nullptr, Dc, Dc, Dc);
    k_gemm<EPI_F32    ><<<gGrid(Dc), 256, 0, stream>>>(axv, sWv[l], vbuf, nullptr, nullptr, Dc, Dc, Dc);
    k_wkv<<<(Bc * Dc + 255) / 256, 256, 0, stream>>>(kbuf, vbuf, tdec + l * Dc, tfirst + l * Dc, rbuf, aatt);
    k_gemm<EPI_ADD    ><<<gGrid(Dc), 256, 0, stream>>>(aatt, sWo[l], x, nullptr, nullptr, Dc, Dc, Dc);
    // ---- channel mixing ----
    k_ln<false><<<Mc, 256, 0, stream>>>(nullptr, x, ln2_g + l * Dc, ln2_b + l * Dc, xx, nullptr);
    k_cm_mix<<<dim3(3, Mc), 256, 0, stream>>>(xx, cm_mu_k + l * Dc, cm_mu_r + l * Dc, axk, axr);
    k_gemm<EPI_RELUSQ ><<<gGrid(Hc), 256, 0, stream>>>(axk, sCk[l], nullptr, kkh, nullptr, Hc, Hc, Dc);
    k_gemm<EPI_SIGMOID><<<gGrid(Dc), 256, 0, stream>>>(axr, sCr[l], r2, nullptr, nullptr, Dc, Dc, Dc);
    k_gemm<EPI_MULADD ><<<gGrid(Dc), 256, 0, stream>>>(kkh, sCv[l], x, nullptr, r2, Dc, Dc, Hc);
  }

  // final LN (straight to bf16) + head projection to logits (non-temporal stores)
  k_ln<true><<<Mc, 256, 0, stream>>>(nullptr, x, lnf_g, lnf_b, nullptr, ahead);
  k_gemm<EPI_F32_NT><<<gGrid(Vpc), 256, 0, stream>>>(ahead, sHead, (float*)d_out, nullptr, nullptr, Vc, Vpc, Dc);
}
